// PanguModel_63007170232397
// MI455X (gfx1250) — matrix operand order repaired and checked by reading, not compiled, not run on hardware
//
#include <hip/hip_runtime.h>
#include <hip/hip_bf16.h>

typedef __attribute__((ext_vector_type(16))) __bf16 v16bf;
typedef __attribute__((ext_vector_type(8)))  float  v8f;
typedef __attribute__((ext_vector_type(4)))  unsigned int u32x4;
typedef int i32x4 __attribute__((vector_size(16)));

#define NW_TOTAL 1024
#define N_TOK    144
#define DIM      192
#define HEADS    6
#define HD       32
#define NPAD     160
#define ZH_WIN   64

// LDS layout (bytes)
#define OFF_X    0        // x  bf16 [144][192] = 55296
#define OFF_Q    55296    // Q  bf16 [144][32]  =  9216
#define OFF_K    64512    // K  bf16 [144][32]  =  9216
#define OFF_VT   73728    // Vt bf16 [32][160]  = 10240
#define OFF_P    83968    // P  bf16 [9][16][160] = 46080
#define OFF_O    130048   // O  bf16 [144][192] = 55296
#define LDS_BYTES 185344
// f32 staging for async x copy aliases OFF_Q..end (110592 B needed, 130048 B available)

// workspace layout (bytes)
#define WS_BIAS   0                       // f32 [64][6][144][144] = 31850496
#define WS_WQKV   31850496                // bf16 packed frags     =   221184
#define WS_WOUT   (31850496 + 221184)     // bf16 packed frags     =    73728

#ifndef __has_builtin
#define __has_builtin(x) 0
#endif

__device__ __forceinline__ v8f wmma_bf16(v16bf a, v16bf b, v8f c) {
  return __builtin_amdgcn_wmma_f32_16x16x32_bf16(false, a, false, b, (short)0, c, false, false);
}

// A fragment (16x32 bf16): lane holds row (lane&15); VGPR0-7 = K kb..kb+15
// contiguous with kb = (lane>>4)*16.  Two 16B loads.
__device__ __forceinline__ v16bf load_a_frag(const __bf16* base, int row_stride_bytes,
                                             int row, int k_off, int lane) {
  const char* p = (const char*)base + (size_t)row * row_stride_bytes
                + (size_t)(k_off + ((lane >> 4) << 4)) * 2;
  union { u32x4 u[2]; v16bf v; } r;
  r.u[0] = *(const u32x4*)p;
  r.u[1] = *(const u32x4*)(p + 16);
  return r.v;
}

// B fragment (32x16 bf16) from pre-packed fragment-major buffer: 32 contiguous bytes.
__device__ __forceinline__ v16bf load_b_frag_packed(const __bf16* pk, int fragid, int lane) {
  const u32x4* p = (const u32x4*)(pk + ((size_t)fragid * 32 + lane) * 16);
  union { u32x4 u[2]; v16bf v; } r;
  r.u[0] = p[0]; r.u[1] = p[1];
  return r.v;
}

// B fragment where logical B[k][n] is stored with n as the row:  base[n][k].
// lane: n = n0 + (lane&15); K = k_off + (lane>>4)*16 .. +15 contiguous.
__device__ __forceinline__ v16bf load_b_frag_rows(const __bf16* base, int row_stride_elems,
                                                  int n0, int k_off, int lane) {
  const __bf16* p = base + (size_t)(n0 + (lane & 15)) * row_stride_elems
                  + k_off + ((lane >> 4) << 4);
  const u32x4* q = (const u32x4*)p;
  union { u32x4 u[2]; v16bf v; } r;
  r.u[0] = q[0]; r.u[1] = q[1];
  return r.v;
}

// ---------------- prep kernel: pack a row-major f32 weight [Ktot][Ntot] into
// WMMA-B fragment-major bf16 (frag = (kk*ntiles + nt), 32 lanes x 16 elems) ----
__global__ void pack_w_kernel(const float* __restrict__ W, __bf16* __restrict__ out,
                              int Ktot, int Ntot) {
  int ntiles = Ntot >> 4;
  int total  = (Ktot >> 5) * ntiles * 32 * 16;
  for (int e = blockIdx.x * blockDim.x + threadIdx.x; e < total;
       e += gridDim.x * blockDim.x) {
    int i16 = e & 15;
    int l   = (e >> 4) & 31;
    int f   = e >> 9;
    int nt  = f % ntiles;
    int kk  = f / ntiles;
    int k = kk * 32 + ((l >> 4) << 4) + i16;
    int n = nt * 16 + (l & 15);
    out[e] = (__bf16)W[(size_t)k * Ntot + n];
  }
}

// ---------------- prep kernel: gather earth-specific bias into [zh][h][n][m] ----
__global__ void bias_pre_kernel(const float* __restrict__ table,
                                const int* __restrict__ pidx,
                                float* __restrict__ outb) {
  const int total = ZH_WIN * HEADS * N_TOK * N_TOK;
  for (int e = blockIdx.x * blockDim.x + threadIdx.x; e < total;
       e += gridDim.x * blockDim.x) {
    int m  = e % N_TOK;
    int n  = (e / N_TOK) % N_TOK;
    int h  = (e / (N_TOK * N_TOK)) % HEADS;
    int zh = e / (N_TOK * N_TOK * HEADS);
    int row = pidx[n * N_TOK + m];
    outb[e] = table[((size_t)row * ZH_WIN + zh) * HEADS + h];
  }
}

// ---------------- fused attention: one workgroup (9 waves) per window ----------
extern "C" __global__ __launch_bounds__(288, 1)
void pangu_attn_fused(const float* __restrict__ x,
                      const float* __restrict__ b_qkv,
                      const float* __restrict__ b_out,
                      const float* __restrict__ bias_pre,
                      const __bf16* __restrict__ wqkv_pk,
                      const __bf16* __restrict__ wout_pk,
                      float* __restrict__ out) {
  extern __shared__ char smem[];
  __bf16* Xs = (__bf16*)(smem + OFF_X);    // [144][192]
  __bf16* Qs = (__bf16*)(smem + OFF_Q);    // [144][32]
  __bf16* Ks = (__bf16*)(smem + OFF_K);    // [144][32]
  __bf16* Vt = (__bf16*)(smem + OFF_VT);   // [32][160]  (V transposed, tokens padded)
  __bf16* Ps = (__bf16*)(smem + OFF_P);    // [9][16][160]
  __bf16* Os = (__bf16*)(smem + OFF_O);    // [144][192]

  const int w    = blockIdx.x;
  const int tid  = threadIdx.x;
  const int lane = tid & 31;
  const int wave = tid >> 5;     // 0..8 -> 16-row strip
  const int zh   = w >> 4;       // bias broadcasts over 16 lon windows
  const int half = lane >> 4;
  const int l15  = lane & 15;
  const int i0   = wave * 16;

  const float* xg = x + (size_t)w * N_TOK * DIM;

#if __has_builtin(__builtin_amdgcn_global_load_async_to_lds_b128)
  // ---- async DMA: stream the f32 window tile straight into LDS (ASYNCcnt) ----
  {
    float* stage = (float*)(smem + OFF_Q);   // aliases Q..O, free until first barrier
#pragma unroll
    for (int it = 0; it < 24; ++it) {
      int idx = it * 288 * 4 + tid * 4;      // 4 floats (b128) per lane per issue
      __builtin_amdgcn_global_load_async_to_lds_b128(
          (__attribute__((address_space(1))) i32x4*)(xg + idx),
          (__attribute__((address_space(3))) i32x4*)(stage + idx), 0, 0);
    }
#if __has_builtin(__builtin_amdgcn_s_wait_asynccnt)
    __builtin_amdgcn_s_wait_asynccnt(0);
#else
    asm volatile("s_wait_asynccnt 0x0" ::: "memory");
#endif
    __syncthreads();
    for (int idx = tid; idx < N_TOK * DIM; idx += 288)
      Xs[idx] = (__bf16)stage[idx];
    __syncthreads();                         // staging region is reused below
  }
#else
  for (int idx = tid; idx < N_TOK * DIM; idx += 288)
    Xs[idx] = (__bf16)xg[idx];
#endif

  // zero padded regions (V tokens 144..159, P cols 144..159)
  if (tid < 512)
    Vt[(tid >> 4) * NPAD + 144 + (tid & 15)] = (__bf16)0.f;
  for (int idx = tid; idx < N_TOK * 16; idx += 288)
    Ps[(idx >> 4) * NPAD + 144 + (idx & 15)] = (__bf16)0.f;
  __syncthreads();

  for (int h = 0; h < HEADS; ++h) {
    // ---- QKV projection for this head: rows i0..i0+15 ----
    v16bf xa[6];
#pragma unroll
    for (int kk = 0; kk < 6; ++kk)
      xa[kk] = load_a_frag(Xs, DIM * 2, i0 + l15, kk * 32, lane);

#pragma unroll
    for (int sec = 0; sec < 3; ++sec) {      // 0=Q 1=K 2=V
#pragma unroll
      for (int sub = 0; sub < 2; ++sub) {
        int nt = sec * 12 + 2 * h + sub;     // column tile in [192][576]
        v8f acc = {};
#pragma unroll
        for (int kk = 0; kk < 6; ++kk)
          acc = wmma_bf16(xa[kk], load_b_frag_packed(wqkv_pk, kk * 36 + nt, lane), acc);
        float bq = b_qkv[sec * 192 + h * 32 + sub * 16 + l15];
#pragma unroll
        for (int r = 0; r < 8; ++r) {
          float v = acc[r] + bq;
          int tok = i0 + r + 8 * half;
          int d   = sub * 16 + l15;
          if (sec == 0)      Qs[tok * HD + d] = (__bf16)v;
          else if (sec == 1) Ks[tok * HD + d] = (__bf16)v;
          else               Vt[d * NPAD + tok] = (__bf16)v;   // store V transposed
        }
      }
    }

    // prefetch next head's bias strip into cache while we sync + do attention
    if (h + 1 < HEADS) {
      const float* bpn = bias_pre + ((size_t)zh * HEADS + h + 1) * N_TOK * N_TOK
                       + (size_t)(i0 + l15) * N_TOK + half * 72;
      __builtin_prefetch(bpn, 0, 1);
      __builtin_prefetch(bpn + 36, 0, 1);
    }
    __syncthreads();

    // ---- S = scale * Q K^T + bias ; softmax over columns ----
    v16bf qa = load_a_frag(Qs, HD * 2, i0 + l15, 0, lane);
    const float scale = 0.17677669529663687f;   // 1/sqrt(32)
    const float* bp = bias_pre + ((size_t)zh * HEADS + h) * N_TOK * N_TOK;
    float sv[9][8];
    float mx[8];
#pragma unroll
    for (int r = 0; r < 8; ++r) mx[r] = -1e30f;
#pragma unroll
    for (int j = 0; j < 9; ++j) {
      v8f zero = {};
      v8f s = wmma_bf16(qa, load_b_frag_rows(Ks, HD, j * 16, 0, lane), zero);
#pragma unroll
      for (int r = 0; r < 8; ++r) {
        int nrow = i0 + r + 8 * half;
        float v = s[r] * scale + bp[nrow * N_TOK + j * 16 + l15];
        sv[j][r] = v;
        mx[r] = fmaxf(mx[r], v);
      }
    }
#pragma unroll
    for (int r = 0; r < 8; ++r) {
#pragma unroll
      for (int m = 1; m < 16; m <<= 1)
        mx[r] = fmaxf(mx[r], __shfl_xor(mx[r], m, 32));
    }
    float sum[8];
#pragma unroll
    for (int r = 0; r < 8; ++r) sum[r] = 0.f;
#pragma unroll
    for (int j = 0; j < 9; ++j)
#pragma unroll
      for (int r = 0; r < 8; ++r) {
        float e = __expf(sv[j][r] - mx[r]);
        sv[j][r] = e;
        sum[r] += e;
      }
#pragma unroll
    for (int r = 0; r < 8; ++r) {
#pragma unroll
      for (int m = 1; m < 16; m <<= 1)
        sum[r] += __shfl_xor(sum[r], m, 32);
      sum[r] = 1.f / sum[r];
    }
    __bf16* pstrip = Ps + wave * 16 * NPAD;
#pragma unroll
    for (int j = 0; j < 9; ++j)
#pragma unroll
      for (int r = 0; r < 8; ++r)
        pstrip[(r + 8 * half) * NPAD + j * 16 + l15] = (__bf16)(sv[j][r] * sum[r]);
    __syncthreads();

    // ---- O strip = P @ V (K padded to 160 with zeros) ----
#pragma unroll
    for (int nt = 0; nt < 2; ++nt) {
      v8f acc = {};
#pragma unroll
      for (int kk = 0; kk < 5; ++kk) {
        v16bf pa = load_a_frag(pstrip, NPAD * 2, l15, kk * 32, lane);
        v16bf vb = load_b_frag_rows(Vt, NPAD, nt * 16, kk * 32, lane);
        acc = wmma_bf16(pa, vb, acc);
      }
#pragma unroll
      for (int r = 0; r < 8; ++r)
        Os[(i0 + r + 8 * half) * DIM + h * HD + nt * 16 + l15] = (__bf16)acc[r];
    }
    __syncthreads();   // protect Q/K/Vt before next head overwrites them
  }

  // ---- out projection: out[w, i0.., :] = O @ w_out + b_out ----
  v16bf oa[6];
#pragma unroll
  for (int kk = 0; kk < 6; ++kk)
    oa[kk] = load_a_frag(Os, DIM * 2, i0 + l15, kk * 32, lane);
#pragma unroll
  for (int nt = 0; nt < 12; ++nt) {
    v8f acc = {};
#pragma unroll
    for (int kk = 0; kk < 6; ++kk)
      acc = wmma_bf16(oa[kk], load_b_frag_packed(wout_pk, kk * 12 + nt, lane), acc);
    float bo = b_out[nt * 16 + l15];
#pragma unroll
    for (int r = 0; r < 8; ++r)
      out[((size_t)w * N_TOK + i0 + r + 8 * half) * DIM + nt * 16 + l15] = acc[r] + bo;
  }
}

extern "C" void kernel_launch(void* const* d_in, const int* in_sizes, int n_in,
                              void* d_out, int out_size, void* d_ws, size_t ws_size,
                              hipStream_t stream) {
  (void)in_sizes; (void)n_in; (void)out_size; (void)ws_size;
  const float* x     = (const float*)d_in[0];
  const float* w_qkv = (const float*)d_in[1];
  const float* b_qkv = (const float*)d_in[2];
  const float* w_out = (const float*)d_in[3];
  const float* b_out = (const float*)d_in[4];
  const float* btab  = (const float*)d_in[5];
  const int*   pidx  = (const int*)d_in[6];

  char*   ws       = (char*)d_ws;
  float*  bias_pre = (float*)(ws + WS_BIAS);
  __bf16* wqkv_pk  = (__bf16*)(ws + WS_WQKV);
  __bf16* wout_pk  = (__bf16*)(ws + WS_WOUT);

  bias_pre_kernel<<<4096, 256, 0, stream>>>(btab, pidx, bias_pre);
  pack_w_kernel<<<128, 256, 0, stream>>>(w_qkv, wqkv_pk, DIM, 3 * DIM);
  pack_w_kernel<<<64, 256, 0, stream>>>(w_out, wout_pk, DIM, DIM);

  (void)hipFuncSetAttribute(reinterpret_cast<const void*>(pangu_attn_fused),
                            hipFuncAttributeMaxDynamicSharedMemorySize, LDS_BYTES);
  pangu_attn_fused<<<NW_TOTAL, 288, LDS_BYTES, stream>>>(
      x, b_qkv, b_out, bias_pre, wqkv_pk, wout_pk, (float*)d_out);
}
